// GLRUCell_53987738910744
// MI455X (gfx1250) — matrix operand order repaired and checked by reading, not compiled, not run on hardware
//
#include <hip/hip_runtime.h>
#include <math.h>

// ---------------------------------------------------------------------------
// GLRU on gfx1250:
//   * bf16 WMMA GEMMs (v_wmma_f32_16x16x32_bf16) with double-buffered
//     GLOBAL_LOAD_ASYNC_TO_LDS_B128 staging (ASYNCcnt pipeline)
//   * channel-parallel diagonal scan
//   * LDS-reduction LayerNorm
// ---------------------------------------------------------------------------

typedef __bf16 bf16_t;
typedef __attribute__((ext_vector_type(16))) __bf16 v16bf;
typedef __attribute__((ext_vector_type(8)))  __bf16 v8bf;
typedef __attribute__((ext_vector_type(8)))  float  v8f;

#define D_DIM 1024
#define B_DIM 16
#define T_DIM 2048
#define M_DIM (B_DIM * T_DIM)   // 32768 rows (b,t flattened)
#define LDS_STRIDE 40           // 32 K-elems + 8 pad (80 B rows -> bank spread)

// round-to-nearest-even f32 -> bf16
__device__ __forceinline__ bf16_t f2bf(float f) {
    unsigned int u = __float_as_uint(f);
    unsigned int r = (u + 0x7FFFu + ((u >> 16) & 1u)) >> 16;
    unsigned short s = (unsigned short)r;
    bf16_t b;
    __builtin_memcpy(&b, &s, sizeof(b));
    return b;
}

// ---- CDNA5 async copy: global -> LDS, 16 bytes, tracked by ASYNCcnt --------
__device__ __forceinline__ void async_copy_b128(unsigned lds_off, const void* gptr) {
    asm volatile("global_load_async_to_lds_b128 %0, %1, off"
                 :: "v"(lds_off), "v"(gptr)
                 : "memory");
}
__device__ __forceinline__ void wait_async0() {
    asm volatile("s_wait_asynccnt 0x0" ::: "memory");
}

__global__ void glru_cvt_f32_bf16(const float* __restrict__ in,
                                  bf16_t* __restrict__ out, int n) {
    int i = blockIdx.x * blockDim.x + threadIdx.x;
    if (i < n) out[i] = f2bf(in[i]);
}

// Stage one K-step: 64x32 A-slab and 64x32 B-slab into LDS.
// 256 threads: thread -> row = tid/4 (0..63), col = (tid%4)*8; 16 B per copy.
__device__ __forceinline__ void stage_pair(const bf16_t* __restrict__ A,
                                           const bf16_t* __restrict__ W,
                                           int m_blk, int n_blk, int k0,
                                           unsigned lds_a, unsigned lds_b, int tid) {
    const int r = tid >> 2;
    const int c = (tid & 3) << 3;
    const unsigned loff = (unsigned)((r * LDS_STRIDE + c) * (int)sizeof(bf16_t));
    async_copy_b128(lds_a + loff, A + (size_t)(m_blk + r) * D_DIM + k0 + c);
    async_copy_b128(lds_b + loff, W + (size_t)(n_blk + r) * D_DIM + k0 + c);
}

// A fragment (16x32 bf16) from LDS: lane<16 -> K {0..7, 16..23}; lane>=16 -> {8..15, 24..31}
__device__ __forceinline__ v16bf a_frag_lds(const bf16_t* sa, int row, int hi) {
    const bf16_t* p = sa + row * LDS_STRIDE + hi * 8;
    v8bf c0 = *reinterpret_cast<const v8bf*>(p);
    v8bf c1 = *reinterpret_cast<const v8bf*>(p + 16);
    v16bf a;
#pragma unroll
    for (int i = 0; i < 8; ++i) { a[i] = c0[i]; a[i + 8] = c1[i]; }
    return a;
}

// B fragment (32x16 bf16, B[k,n]=W[n,k]): lane holds N=lane%16, lane<16 -> K {0..7, 16..23}; lane>=16 -> {8..15, 24..31}
__device__ __forceinline__ v16bf b_frag_lds(const bf16_t* sb, int row, int hi) {
    const bf16_t* p = sb + row * LDS_STRIDE + hi * 8;
    v8bf c0 = *reinterpret_cast<const v8bf*>(p);
    v8bf c1 = *reinterpret_cast<const v8bf*>(p + 16);
    v16bf b;
#pragma unroll
    for (int i = 0; i < 8; ++i) { b[i] = c0[i]; b[i + 8] = c1[i]; }
    return b;
}

__device__ __forceinline__ void wmma_step(const bf16_t* sa, const bf16_t* sb,
                                          int ar, int br, int hi,
                                          v8f& acc0, v8f& acc1) {
    v16bf a  = a_frag_lds(sa, ar, hi);
    v16bf b0 = b_frag_lds(sb, br, hi);
    v16bf b1 = b_frag_lds(sb, br + 16, hi);
    acc0 = __builtin_amdgcn_wmma_f32_16x16x32_bf16(false, a, false, b0, (short)0, acc0, false, false);
    acc1 = __builtin_amdgcn_wmma_f32_16x16x32_bf16(false, a, false, b1, (short)0, acc1, false, false);
}

// Shared tile pipeline: computes a 64x64 block tile of  A[M,K] @ W[N,K]^T .
// 8 waves: wave&3 -> 16-row M strip, wave>>2 -> 32-col N strip (2 accumulators).
#define GEMM_TILE_BODY(Amat, Wmat)                                                     \
    __shared__ alignas(16) bf16_t lds_a[2][64 * LDS_STRIDE];                           \
    __shared__ alignas(16) bf16_t lds_b[2][64 * LDS_STRIDE];                           \
    const int tid  = threadIdx.x;                                                      \
    const int lane = tid & 31;                                                         \
    const int wave = tid >> 5;                                                         \
    const int hi   = lane >> 4;                                                        \
    const int m_blk = blockIdx.y * 64;                                                 \
    const int n_blk = blockIdx.x * 64;                                                 \
    const int ar = (wave & 3) * 16 + (lane & 15);                                      \
    const int br = (wave >> 2) * 32 + (lane & 15);                                     \
    const unsigned la0 = (unsigned)(size_t)&lds_a[0][0];                               \
    const unsigned la1 = (unsigned)(size_t)&lds_a[1][0];                               \
    const unsigned lb0 = (unsigned)(size_t)&lds_b[0][0];                               \
    const unsigned lb1 = (unsigned)(size_t)&lds_b[1][0];                               \
    v8f acc0 = {}; v8f acc1 = {};                                                      \
    stage_pair(Amat, Wmat, m_blk, n_blk, 0, la0, lb0, tid);                            \
    for (int k0 = 0; k0 < D_DIM; k0 += 64) {                                           \
        wait_async0();                                                                 \
        __syncthreads();                                                               \
        if (k0 + 32 < D_DIM)                                                           \
            stage_pair(Amat, Wmat, m_blk, n_blk, k0 + 32, la1, lb1, tid);              \
        wmma_step(lds_a[0], lds_b[0], ar, br, hi, acc0, acc1);                         \
        wait_async0();                                                                 \
        __syncthreads();                                                               \
        if (k0 + 64 < D_DIM)                                                           \
            stage_pair(Amat, Wmat, m_blk, n_blk, k0 + 64, la0, lb0, tid);              \
        wmma_step(lds_a[1], lds_b[1], ar, br, hi, acc0, acc1);                         \
    }

// GEMM 1: proj = Xb @ Winb^T over N = 2*D columns.
// Epilogue: cols < D -> u = sigmoid(proj) (f32); cols >= D -> vx as bf16.
__global__ void __launch_bounds__(256) glru_gemm_proj(
    const bf16_t* __restrict__ Xb,     // [M, D] bf16
    const bf16_t* __restrict__ Winb,   // [2D, D] bf16
    float* __restrict__ u_out,         // [M, D] f32
    bf16_t* __restrict__ vx_out)       // [M, D] bf16
{
    GEMM_TILE_BODY(Xb, Winb)

    const int col = lane & 15;
#pragma unroll
    for (int r = 0; r < 8; ++r) {
        const int m  = m_blk + (wave & 3) * 16 + r + hi * 8;
        const int na = n_blk + (wave >> 2) * 32 + col;
        const int nb = na + 16;
        const float va = acc0[r], vb = acc1[r];
        if (na < D_DIM) u_out[(size_t)m * D_DIM + na] = 1.0f / (1.0f + expf(-va));
        else            vx_out[(size_t)m * D_DIM + (na - D_DIM)] = f2bf(va);
        if (nb < D_DIM) u_out[(size_t)m * D_DIM + nb] = 1.0f / (1.0f + expf(-vb));
        else            vx_out[(size_t)m * D_DIM + (nb - D_DIM)] = f2bf(vb);
    }
}

// GEMM 2: cand = vx @ W_state^T (N = D), f32 out.
__global__ void __launch_bounds__(256) glru_gemm_cand(
    const bf16_t* __restrict__ Vxb,    // [M, D] bf16
    const bf16_t* __restrict__ Wsb,    // [D, D] bf16
    float* __restrict__ cand_out)      // [M, D] f32
{
    GEMM_TILE_BODY(Vxb, Wsb)

    const int col = lane & 15;
#pragma unroll
    for (int r = 0; r < 8; ++r) {
        const int m = m_blk + (wave & 3) * 16 + r + hi * 8;
        const int n = n_blk + (wave >> 2) * 32 + col;
        cand_out[(size_t)m * D_DIM + n]      = acc0[r];
        cand_out[(size_t)m * D_DIM + n + 16] = acc1[r];
    }
}

// Diagonal linear recurrence: one thread per (b,d) channel, sequential in t.
__global__ void glru_scan(const float* __restrict__ u,
                          const float* __restrict__ cand,
                          float* __restrict__ y)
{
    const int idx = blockIdx.x * blockDim.x + threadIdx.x;   // b*D + d
    const int b = idx / D_DIM;
    const int d = idx % D_DIM;
    size_t off = (size_t)b * T_DIM * D_DIM + d;
    float h = 0.0f;
    for (int t = 0; t < T_DIM; ++t, off += D_DIM) {
        const float ut = u[off];
        const float ct = cand[off];
        h = ut * h + (1.0f - ut) * ct;
        y[off] = h;
    }
}

// Row LayerNorm over D, in place on y (d_out). One 256-thread block per row.
__global__ void __launch_bounds__(256) glru_layernorm(
    float* __restrict__ y,
    const float* __restrict__ gamma,
    const float* __restrict__ beta)
{
    __shared__ float red_s[256];
    __shared__ float red_q[256];
    const int row = blockIdx.x;
    const int tid = threadIdx.x;
    float* p = y + (size_t)row * D_DIM;

    float vals[4];
    float s = 0.0f, q = 0.0f;
#pragma unroll
    for (int i = 0; i < 4; ++i) {
        const float v = p[tid + i * 256];
        vals[i] = v;
        s += v;
        q += v * v;
    }
    red_s[tid] = s; red_q[tid] = q;
    __syncthreads();
#pragma unroll
    for (int o = 128; o > 0; o >>= 1) {
        if (tid < o) { red_s[tid] += red_s[tid + o]; red_q[tid] += red_q[tid + o]; }
        __syncthreads();
    }
    const float mu  = red_s[0] * (1.0f / D_DIM);
    const float var = red_q[0] * (1.0f / D_DIM) - mu * mu;
    const float inv = rsqrtf(var + 1e-5f);
#pragma unroll
    for (int i = 0; i < 4; ++i) {
        const int d = tid + i * 256;
        p[d] = (vals[i] - mu) * inv * gamma[d] + beta[d];
    }
}

extern "C" void kernel_launch(void* const* d_in, const int* in_sizes, int n_in,
                              void* d_out, int out_size, void* d_ws, size_t ws_size,
                              hipStream_t stream) {
    (void)in_sizes; (void)n_in; (void)out_size; (void)ws_size;
    const float* x     = (const float*)d_in[0];   // [B,T,D]
    const float* W_in  = (const float*)d_in[1];   // [2D,D]
    const float* W_st  = (const float*)d_in[2];   // [D,D]
    const float* gamma = (const float*)d_in[3];   // [D]
    const float* beta  = (const float*)d_in[4];   // [D]

    size_t off = 0;
    auto carve = [&](size_t bytes) -> void* {
        void* p = (char*)d_ws + off;
        off += (bytes + 255) & ~(size_t)255;
        return p;
    };
    const size_t MD = (size_t)M_DIM * D_DIM;                 // 33,554,432
    bf16_t* xb   = (bf16_t*)carve(MD * sizeof(bf16_t));                        //  64 MB
    bf16_t* winb = (bf16_t*)carve((size_t)2 * D_DIM * D_DIM * sizeof(bf16_t)); //   4 MB
    bf16_t* wsb  = (bf16_t*)carve((size_t)D_DIM * D_DIM * sizeof(bf16_t));     //   2 MB
    float*  u    = (float*) carve(MD * sizeof(float));                         // 128 MB
    bf16_t* vxb  = (bf16_t*)carve(MD * sizeof(bf16_t));                        //  64 MB
    float*  cand = (float*) carve(MD * sizeof(float));                         // 128 MB
    float*  y    = (float*)d_out;

    // 1) down-convert inputs/weights to bf16 (weights stay L2-resident)
    glru_cvt_f32_bf16<<<(unsigned)(MD / 256), 256, 0, stream>>>(x, xb, (int)MD);
    glru_cvt_f32_bf16<<<(2 * D_DIM * D_DIM) / 256, 256, 0, stream>>>(W_in, winb, 2 * D_DIM * D_DIM);
    glru_cvt_f32_bf16<<<(D_DIM * D_DIM) / 256, 256, 0, stream>>>(W_st, wsb, D_DIM * D_DIM);

    // 2) proj GEMM (M=32768, N=2048, K=1024), fused sigmoid / bf16 requant
    glru_gemm_proj<<<dim3((2 * D_DIM) / 64, M_DIM / 64), 256, 0, stream>>>(xb, winb, u, vxb);

    // 3) cand GEMM (M=32768, N=1024, K=1024)
    glru_gemm_cand<<<dim3(D_DIM / 64, M_DIM / 64), 256, 0, stream>>>(vxb, wsb, cand);

    // 4) channel-parallel recurrence (16384 threads, T=2048 steps)
    glru_scan<<<(B_DIM * D_DIM) / 256, 256, 0, stream>>>(u, cand, y);

    // 5) in-place LayerNorm per (b,t) row
    glru_layernorm<<<M_DIM, 256, 0, stream>>>(y, gamma, beta);
}
